// TemporalGCN1_28724741276161
// MI455X (gfx1250) — matrix operand order repaired and checked by reading, not compiled, not run on hardware
//
#include <hip/hip_runtime.h>
#include <hip/hip_bf16.h>
#include <math.h>

#define T_STEPS 8
#define NN 10000
#define EE 160000
#define FIN 16
#define HH 4
#define CC 32
#define DD 128
#define GG 384   // 3*D GRU gate width

typedef _Float16 v16h __attribute__((ext_vector_type(16)));
typedef _Float16 v8h  __attribute__((ext_vector_type(8)));
typedef float    v8f  __attribute__((ext_vector_type(8)));

// ---------------- helpers ----------------
__device__ __forceinline__ void atomicMaxFloat(float* addr, float val) {
  // classic sign-split trick: works with init = -INF, handles negatives
  if (val >= 0.0f) atomicMax((int*)addr, __float_as_int(val));
  else             atomicMin((unsigned int*)addr, __float_as_uint(val));
}

// ---------------- phase 1: per-timestep TransformerConv(16->128,H=4) ----------------

// One block (128 threads) per node: q,k,v projections + skip init + softmax-state init.
__global__ __launch_bounds__(128)
void k_proj1(const float* __restrict__ x_seq, int t,
             const float* __restrict__ Wq, const float* __restrict__ bq,
             const float* __restrict__ Wk, const float* __restrict__ bk,
             const float* __restrict__ Wv, const float* __restrict__ bv,
             const float* __restrict__ Ws, const float* __restrict__ bs,
             float* __restrict__ q, float* __restrict__ k, float* __restrict__ v,
             float* __restrict__ accum, float* __restrict__ amax, float* __restrict__ denom) {
  __shared__ float xs[FIN];
  const int n = blockIdx.x;
  const int j = threadIdx.x;
  if (j < FIN) xs[j] = x_seq[(size_t)(t * NN + n) * FIN + j];
  if (j < HH) { amax[n * HH + j] = -__builtin_inff(); denom[n * HH + j] = 0.0f; }
  __syncthreads();
  float qv = bq[j], kv = bk[j], vv = bv[j], sv = bs[j];
#pragma unroll
  for (int i = 0; i < FIN; ++i) {
    const float xi = xs[i];
    qv += xi * Wq[i * DD + j];
    kv += xi * Wk[i * DD + j];
    vv += xi * Wv[i * DD + j];
    sv += xi * Ws[i * DD + j];
  }
  q[n * DD + j] = qv; k[n * DD + j] = kv; v[n * DD + j] = vv;
  accum[n * DD + j] = sv;   // skip connection seeds the output accumulator
}

// per (edge, head): attention logit + segment max (atomic)
__global__ __launch_bounds__(256)
void k_edge_alpha(const int* __restrict__ ei, const float* __restrict__ ea,
                  const float* __restrict__ q, const float* __restrict__ k,
                  const float* __restrict__ We, float* __restrict__ alpha,
                  float* __restrict__ amax) {
  const int tid = blockIdx.x * 256 + threadIdx.x;   // E*H threads
  const int h = tid & (HH - 1);
  const int e = tid >> 2;
  const int src = ei[e];
  const int dst = ei[EE + e];
  const float a = ea[e];
  const float4* qp = (const float4*)(q + (size_t)dst * DD + h * CC);
  const float4* kp = (const float4*)(k + (size_t)src * DD + h * CC);
  const float4* wp = (const float4*)(We + h * CC);
  float s = 0.0f;
#pragma unroll
  for (int i = 0; i < CC / 4; ++i) {
    const float4 qq = qp[i], kk = kp[i], ww = wp[i];
    s += qq.x * (kk.x + a * ww.x) + qq.y * (kk.y + a * ww.y)
       + qq.z * (kk.z + a * ww.z) + qq.w * (kk.w + a * ww.w);
  }
  s *= 0.17677669529663687f;  // 1/sqrt(32)
  alpha[tid] = s;
  atomicMaxFloat(&amax[dst * HH + h], s);
}

// per (edge, head): exp(alpha - max) + segment sum (atomic)
__global__ __launch_bounds__(256)
void k_edge_expsum(const int* __restrict__ ei, float* __restrict__ alpha,
                   const float* __restrict__ amax, float* __restrict__ denom) {
  const int tid = blockIdx.x * 256 + threadIdx.x;
  const int h = tid & (HH - 1);
  const int e = tid >> 2;
  const int dst = ei[EE + e];
  float m = amax[dst * HH + h];
  if (!isfinite(m)) m = 0.0f;               // reference: where(isfinite, amax, 0)
  const float ex = __expf(alpha[tid] - m);
  alpha[tid] = ex;
  atomicAdd(&denom[dst * HH + h], ex);
}

// per (edge, channel): weighted message scatter-add
__global__ __launch_bounds__(256)
void k_edge_msg(const int* __restrict__ ei, const float* __restrict__ ea,
                const float* __restrict__ v, const float* __restrict__ We,
                const float* __restrict__ alpha, const float* __restrict__ denom,
                float* __restrict__ accum) {
  const int tid = blockIdx.x * 256 + threadIdx.x;   // E*128 threads
  const int c = tid & (DD - 1);
  const int e = tid >> 7;
  const int h = c >> 5;
  const int src = ei[e];
  const int dst = ei[EE + e];
  const float a = ea[e];
  const float w = alpha[e * HH + h] / (denom[dst * HH + h] + 1e-16f);
  const float val = (v[(size_t)src * DD + c] + a * We[c]) * w;
  atomicAdd(&accum[(size_t)dst * DD + c], val);
}

// leaky_relu + f16 cast into the GNN time-stack (GRU input)
__global__ __launch_bounds__(256)
void k_finalize1(const float* __restrict__ accum, _Float16* __restrict__ gnn16) {
  const int i = blockIdx.x * 256 + threadIdx.x;     // N*128 threads
  const float x = accum[i];
  const float y = x > 0.0f ? x : 0.01f * x;
  gnn16[i] = (_Float16)y;
}

// ---------------- phase 2: GRU(128) with WMMA ----------------

// Swizzle f32 [384,128] weights into WMMA B-fragment order (f16).
// B[k][g] = W[g][k]; fragment (nt,kt): lane holds col g = nt*16 + (lane&15),
// half j holds k = kt*32 + (lane>>4)*16 + j.
__global__ __launch_bounds__(256)
void k_prep_gru(const float* __restrict__ Wih, const float* __restrict__ Whh,
                _Float16* __restrict__ fragIH, _Float16* __restrict__ fragHH) {
  const int tid = blockIdx.x * 256 + threadIdx.x;   // 2 * 384*128 threads
  const int which = tid >= (GG * DD);
  const int idx = which ? tid - GG * DD : tid;
  const int j = idx & 15;
  const int lane = (idx >> 4) & 31;
  const int kt = (idx >> 9) & 3;
  const int nt = idx >> 11;
  const int g = nt * 16 + (lane & 15);
  const int kk = kt * 32 + (lane >> 4) * 16 + j;
  const float w = which ? Whh[g * DD + kk] : Wih[g * DD + kk];
  (which ? fragHH : fragIH)[idx] = (_Float16)w;
}

__global__ __launch_bounds__(256)
void k_init_h(float* __restrict__ h, _Float16* __restrict__ h16) {
  const int i = blockIdx.x * 256 + threadIdx.x;     // N*128
  h[i] = 0.0f; h16[i] = (_Float16)0.0f;
}

// C = A[N,128](f16 row-major) x Bfrag -> out[N,384] f32.
// grid(625, 6), block 128 (4 waves); wave -> one 16x16 tile; K = 4 x 32.
__global__ __launch_bounds__(128)
void k_gemm_wmma(const _Float16* __restrict__ A, const _Float16* __restrict__ Bfrag,
                 float* __restrict__ out) {
  const int lane = threadIdx.x & 31;
  const int wave = threadIdx.x >> 5;
  const int mtile = blockIdx.x;
  const int nt = blockIdx.y * 4 + wave;
  const int grp = lane >> 4;       // lane group selects K 16-block (A and B)
  const int ml = lane & 15;
  const _Float16* arow = A + (size_t)(mtile * 16 + ml) * DD;
  v8f acc = {};
#pragma unroll
  for (int kt = 0; kt < 4; ++kt) {
    // A 16x32 f16 layout: halves 0..15 -> K = kt*32 + grp*16 + j (matches B fragment)
    const v8h lo = *(const v8h*)(arow + kt * 32 + grp * 16);
    const v8h hi = *(const v8h*)(arow + kt * 32 + grp * 16 + 8);
    v16h a;
#pragma unroll
    for (int i = 0; i < 8; ++i) { a[i] = lo[i]; a[8 + i] = hi[i]; }
    const v16h b = *(const v16h*)(Bfrag + ((size_t)(nt * 4 + kt) * 32 + lane) * 16);
    acc = __builtin_amdgcn_wmma_f32_16x16x32_f16(false, a, false, b, (short)0, acc,
                                                 false, false);
  }
  // D layout: VGPR i -> M = grp*8 + i, N = ml
  float* orow = out + (size_t)(mtile * 16 + grp * 8) * GG + nt * 16 + ml;
#pragma unroll
  for (int i = 0; i < 8; ++i) orow[(size_t)i * GG] = acc[i];
}

// GRU gate math (adds biases; updates f32 state + f16 copy for next WMMA step)
__global__ __launch_bounds__(256)
void k_gru_gate(const float* __restrict__ gi, const float* __restrict__ gh,
                const float* __restrict__ bih, const float* __restrict__ bhh,
                float* __restrict__ h, _Float16* __restrict__ h16) {
  const int tid = blockIdx.x * 256 + threadIdx.x;   // N*128
  const int n = tid >> 7, d = tid & (DD - 1);
  const float* gin = gi + (size_t)n * GG;
  const float* ghn = gh + (size_t)n * GG;
  const float ir = gin[d]           + bih[d],            hr = ghn[d]           + bhh[d];
  const float iz = gin[DD + d]      + bih[DD + d],       hz = ghn[DD + d]      + bhh[DD + d];
  const float in_ = gin[2 * DD + d] + bih[2 * DD + d],   hn = ghn[2 * DD + d]  + bhh[2 * DD + d];
  const float r = 1.0f / (1.0f + __expf(-(ir + hr)));
  const float z = 1.0f / (1.0f + __expf(-(iz + hz)));
  const float nc = tanhf(in_ + r * hn);
  const float ho = h[tid];
  const float hv = (1.0f - z) * nc + z * ho;
  h[tid] = hv;
  h16[tid] = (_Float16)hv;
}

// ---------------- phase 3: output TransformerConv(128->1, H=1) ----------------

__global__ __launch_bounds__(256)
void k_proj2(const float* __restrict__ h,
             const float* __restrict__ Wq, const float* __restrict__ bq,
             const float* __restrict__ Wk, const float* __restrict__ bk,
             const float* __restrict__ Wv, const float* __restrict__ bv,
             const float* __restrict__ Ws, const float* __restrict__ bs,
             float* __restrict__ q2, float* __restrict__ k2, float* __restrict__ v2,
             float* __restrict__ dout, float* __restrict__ amax2, float* __restrict__ denom2) {
  const int n = blockIdx.x * 256 + threadIdx.x;
  if (n >= NN) return;
  const float4* hp = (const float4*)(h + (size_t)n * DD);
  float qv = 0.f, kv = 0.f, vv = 0.f, sv = 0.f;
#pragma unroll 8
  for (int i = 0; i < DD / 4; ++i) {
    const float4 hh = hp[i];
    const float4 wq = ((const float4*)Wq)[i];
    const float4 wk = ((const float4*)Wk)[i];
    const float4 wv = ((const float4*)Wv)[i];
    const float4 ws = ((const float4*)Ws)[i];
    qv += hh.x * wq.x + hh.y * wq.y + hh.z * wq.z + hh.w * wq.w;
    kv += hh.x * wk.x + hh.y * wk.y + hh.z * wk.z + hh.w * wk.w;
    vv += hh.x * wv.x + hh.y * wv.y + hh.z * wv.z + hh.w * wv.w;
    sv += hh.x * ws.x + hh.y * ws.y + hh.z * ws.z + hh.w * ws.w;
  }
  q2[n] = qv + bq[0]; k2[n] = kv + bk[0]; v2[n] = vv + bv[0];
  dout[n] = sv + bs[0];            // skip seeds output
  amax2[n] = -__builtin_inff(); denom2[n] = 0.0f;
}

__global__ __launch_bounds__(256)
void k_edge2_alpha(const int* __restrict__ ei, const float* __restrict__ ea,
                   const float* __restrict__ q2, const float* __restrict__ k2,
                   const float* __restrict__ We, float* __restrict__ alpha2,
                   float* __restrict__ amax2) {
  const int e = blockIdx.x * 256 + threadIdx.x;     // E threads
  const int src = ei[e], dst = ei[EE + e];
  const float s = q2[dst] * (k2[src] + ea[e] * We[0]);   // C=1 -> /sqrt(1)
  alpha2[e] = s;
  atomicMaxFloat(&amax2[dst], s);
}

__global__ __launch_bounds__(256)
void k_edge2_expsum(const int* __restrict__ ei, float* __restrict__ alpha2,
                    const float* __restrict__ amax2, float* __restrict__ denom2) {
  const int e = blockIdx.x * 256 + threadIdx.x;
  const int dst = ei[EE + e];
  float m = amax2[dst];
  if (!isfinite(m)) m = 0.0f;
  const float ex = __expf(alpha2[e] - m);
  alpha2[e] = ex;
  atomicAdd(&denom2[dst], ex);
}

__global__ __launch_bounds__(256)
void k_edge2_msg(const int* __restrict__ ei, const float* __restrict__ ea,
                 const float* __restrict__ v2, const float* __restrict__ We,
                 const float* __restrict__ alpha2, const float* __restrict__ denom2,
                 float* __restrict__ dout) {
  const int e = blockIdx.x * 256 + threadIdx.x;
  const int src = ei[e], dst = ei[EE + e];
  const float w = alpha2[e] / (denom2[dst] + 1e-16f);
  atomicAdd(&dout[dst], (v2[src] + ea[e] * We[0]) * w);
}

// ---------------- driver ----------------
extern "C" void kernel_launch(void* const* d_in, const int* in_sizes, int n_in,
                              void* d_out, int out_size, void* d_ws, size_t ws_size,
                              hipStream_t stream) {
  (void)in_sizes; (void)n_in; (void)out_size; (void)ws_size;
  const float* x_seq  = (const float*)d_in[0];
  const int*   ei     = (const int*)d_in[1];     // [T,2,E]
  const float* ea     = (const float*)d_in[2];   // [T,E,1]
  const float *Wq1 = (const float*)d_in[3],  *bq1 = (const float*)d_in[4];
  const float *Wk1 = (const float*)d_in[5],  *bk1 = (const float*)d_in[6];
  const float *Wv1 = (const float*)d_in[7],  *bv1 = (const float*)d_in[8];
  const float *We1 = (const float*)d_in[9];
  const float *Ws1 = (const float*)d_in[10], *bs1 = (const float*)d_in[11];
  const float *Wih = (const float*)d_in[12], *Whh = (const float*)d_in[13];
  const float *bih = (const float*)d_in[14], *bhh = (const float*)d_in[15];
  const float *Wq2 = (const float*)d_in[16], *bq2 = (const float*)d_in[17];
  const float *Wk2 = (const float*)d_in[18], *bk2 = (const float*)d_in[19];
  const float *Wv2 = (const float*)d_in[20], *bv2 = (const float*)d_in[21];
  const float *We2 = (const float*)d_in[22];
  const float *Ws2 = (const float*)d_in[23], *bs2 = (const float*)d_in[24];
  float* dout = (float*)d_out;

  // workspace layout (all chunk sizes are multiples of 64 bytes)
  char* w = (char*)d_ws;
  float* q      = (float*)w; w += (size_t)NN * DD * 4;
  float* k      = (float*)w; w += (size_t)NN * DD * 4;
  float* v      = (float*)w; w += (size_t)NN * DD * 4;
  float* accum  = (float*)w; w += (size_t)NN * DD * 4;
  float* amax   = (float*)w; w += (size_t)NN * HH * 4;
  float* denom  = (float*)w; w += (size_t)NN * HH * 4;
  float* alphab = (float*)w; w += (size_t)EE * HH * 4;
  _Float16* gnn16  = (_Float16*)w; w += (size_t)T_STEPS * NN * DD * 2;
  _Float16* fragIH = (_Float16*)w; w += (size_t)GG * DD * 2;
  _Float16* fragHH = (_Float16*)w; w += (size_t)GG * DD * 2;
  float* gi     = (float*)w; w += (size_t)NN * GG * 4;
  float* gh     = (float*)w; w += (size_t)NN * GG * 4;
  float* hbuf   = (float*)w; w += (size_t)NN * DD * 4;
  _Float16* h16 = (_Float16*)w; w += (size_t)NN * DD * 2;
  float* q2     = (float*)w; w += (size_t)NN * 4;
  float* k2     = (float*)w; w += (size_t)NN * 4;
  float* v2     = (float*)w; w += (size_t)NN * 4;
  float* amax2  = (float*)w; w += (size_t)NN * 4;
  float* denom2 = (float*)w; w += (size_t)NN * 4;
  float* alpha2 = (float*)w; w += (size_t)EE * 4;

  // phase 1: GNN per timestep
  for (int t = 0; t < T_STEPS; ++t) {
    const int* eit = ei + (size_t)t * 2 * EE;
    const float* eat = ea + (size_t)t * EE;
    k_proj1<<<NN, 128, 0, stream>>>(x_seq, t, Wq1, bq1, Wk1, bk1, Wv1, bv1, Ws1, bs1,
                                    q, k, v, accum, amax, denom);
    k_edge_alpha<<<(EE * HH) / 256, 256, 0, stream>>>(eit, eat, q, k, We1, alphab, amax);
    k_edge_expsum<<<(EE * HH) / 256, 256, 0, stream>>>(eit, alphab, amax, denom);
    k_edge_msg<<<(EE * DD) / 256, 256, 0, stream>>>(eit, eat, v, We1, alphab, denom, accum);
    k_finalize1<<<(NN * DD) / 256, 256, 0, stream>>>(accum, gnn16 + (size_t)t * NN * DD);
  }

  // phase 2: GRU with WMMA
  k_prep_gru<<<(2 * GG * DD) / 256, 256, 0, stream>>>(Wih, Whh, fragIH, fragHH);
  k_init_h<<<(NN * DD) / 256, 256, 0, stream>>>(hbuf, h16);
  for (int t = 0; t < T_STEPS; ++t) {
    k_gemm_wmma<<<dim3(NN / 16, (GG / 16) / 4), 128, 0, stream>>>(
        gnn16 + (size_t)t * NN * DD, fragIH, gi);
    k_gemm_wmma<<<dim3(NN / 16, (GG / 16) / 4), 128, 0, stream>>>(h16, fragHH, gh);
    k_gru_gate<<<(NN * DD) / 256, 256, 0, stream>>>(gi, gh, bih, bhh, hbuf, h16);
  }

  // phase 3: output conv on the last graph
  const int* eiL = ei + (size_t)(T_STEPS - 1) * 2 * EE;
  const float* eaL = ea + (size_t)(T_STEPS - 1) * EE;
  k_proj2<<<(NN + 255) / 256, 256, 0, stream>>>(hbuf, Wq2, bq2, Wk2, bk2, Wv2, bv2,
                                                Ws2, bs2, q2, k2, v2, dout, amax2, denom2);
  k_edge2_alpha<<<EE / 256, 256, 0, stream>>>(eiL, eaL, q2, k2, We2, alpha2, amax2);
  k_edge2_expsum<<<EE / 256, 256, 0, stream>>>(eiL, alpha2, amax2, denom2);
  k_edge2_msg<<<EE / 256, 256, 0, stream>>>(eiL, eaL, v2, We2, alpha2, denom2, dout);
}
